// Network_64201171140962
// MI455X (gfx1250) — compile-verified, matrix operand order checked by reading, not run on hardware
//
#include <hip/hip_runtime.h>
#include <hip/hip_bf16.h>
#include <math.h>

// ---------------------------------------------------------------------------
// PointNet++ style network for MI455X (gfx1250, wave32, WMMA).
//
// Key optimization: output depends only on point 0 of each batch through the
// final FP0/fc/head chain, so those stages run on 4 rows only (exact).
// All SA/FP MLP layers run as f16-input/f32-accumulate WMMA GEMMs
// (v_wmma_f32_16x16x32_f16), one wave per 16x16 output tile.
// ---------------------------------------------------------------------------

#define BN_SCALE_C 0.9999950000374997f   // 1/sqrt(1+1e-5)

typedef __attribute__((ext_vector_type(16))) _Float16 v16h;
typedef __attribute__((ext_vector_type(8)))  float    v8f;

// ---------------------------------------------------------------------------
// 0. Prep: pcs[:,0,:3]=cp, pcs[:,0,3]=1 ; split xyz (B,N,3) / feats (B,N,4)
// ---------------------------------------------------------------------------
__global__ void prep_kernel(const float* __restrict__ pcs, const float* __restrict__ cp,
                            float* __restrict__ xyz, float* __restrict__ feats) {
  int i = blockIdx.x * blockDim.x + threadIdx.x;           // B*N
  if (i >= 4 * 16384) return;
  int b = i >> 14, p = i & 16383;
  const float* s = pcs + (size_t)i * 4;
  float x = s[0], y = s[1], z = s[2], w = s[3];
  if (p == 0) { x = cp[b * 3]; y = cp[b * 3 + 1]; z = cp[b * 3 + 2]; w = 1.0f; }
  xyz[(size_t)i * 3 + 0] = x; xyz[(size_t)i * 3 + 1] = y; xyz[(size_t)i * 3 + 2] = z;
  feats[(size_t)i * 4 + 0] = x; feats[(size_t)i * 4 + 1] = y;
  feats[(size_t)i * 4 + 2] = z; feats[(size_t)i * 4 + 3] = w;
}

// ---------------------------------------------------------------------------
// 1. Weight -> WMMA B-fragment conversion (fp32 CinxCout -> f16 fragments,
//    K zero-padded to Kp). Fragment linear index:
//    ((kChunk*tilesN + nTile)*32 + lane)*16 + j
//    lane<16 : col = nTile*16+lane,      halves j<8 -> K=c*32+j,  j>=8 -> K=c*32+16+(j-8)
//    lane>=16: col = nTile*16+(lane-16), halves j<8 -> K=c*32+8+j, j>=8 -> K=c*32+24+(j-8)
// ---------------------------------------------------------------------------
__global__ void wfrag_kernel(const float* __restrict__ W, _Float16* __restrict__ F,
                             int Cin, int Cout, int Kp) {
  int total = Kp * Cout;                                   // == total halves
  int i = blockIdx.x * blockDim.x + threadIdx.x;
  if (i >= total) return;
  int tilesN = Cout >> 4;
  int j    = i & 15;
  int lane = (i >> 4) & 31;
  int r    = i >> 9;
  int nt   = r % tilesN;
  int c    = r / tilesN;
  int n = nt * 16 + (lane & 15);
  int k = c * 32 + ((lane >= 16) ? 8 : 0) + ((j < 8) ? j : (j + 8));
  F[i] = (k < Cin) ? (_Float16)W[(size_t)k * Cout + n] : (_Float16)0.0f;
}

// ---------------------------------------------------------------------------
// 2. WMMA GEMM:  Out[M,Nout] = relu(BN_SCALE * A[M,Kp] @ Wfrag)
//    One wave per 16x16 tile; K stepped by 32 via v_wmma_f32_16x16x32_f16.
// ---------------------------------------------------------------------------
__global__ void __launch_bounds__(256)
gemm_wmma_relu(const float* __restrict__ A, const _Float16* __restrict__ Wf,
               float* __restrict__ Out, int M, int Kp, int Nout) {
  int tilesN = Nout >> 4;
  int totalTiles = (M >> 4) * tilesN;
  int wid = (blockIdx.x << 3) + (threadIdx.x >> 5);
  if (wid >= totalTiles) return;
  int lane = threadIdx.x & 31;
  int mt = wid / tilesN;
  int nt = wid - mt * tilesN;

  int half_ = lane >> 4;          // 0: K-halves {0..7,16..23}, 1: {8..15,24..31}
  int lm    = lane & 15;

  const float* arow = A + (size_t)(mt * 16 + lm) * Kp + (half_ << 3);
  const _Float16* wf = Wf + (size_t)nt * 512 + lane * 16;
  size_t wstride = (size_t)tilesN * 512;

  v8f acc = {};
  int chunks = Kp >> 5;
  for (int c = 0; c < chunks; ++c) {
    const float* p = arow + (c << 5);
    __builtin_prefetch(p + 32, 0, 0);                      // global_prefetch_b8
    v16h a;
#pragma unroll
    for (int j = 0; j < 8; ++j) a[j]     = (_Float16)p[j];
#pragma unroll
    for (int j = 0; j < 8; ++j) a[8 + j] = (_Float16)p[16 + j];
    v16h b = *(const v16h*)wf;
    wf += wstride;
    acc = __builtin_amdgcn_wmma_f32_16x16x32_f16(
        /*neg_a=*/false, a, /*neg_b=*/false, b,
        /*c_mod=*/(short)0, acc, /*reuse_a=*/false, /*reuse_b=*/false);
  }

  // C/D layout: VGPR i, lanes 0-15 -> M=i ; lanes 16-31 -> M=8+i ; N = lane&15
  float* orow = Out + (size_t)(mt * 16 + (half_ << 3)) * Nout + nt * 16 + lm;
#pragma unroll
  for (int i = 0; i < 8; ++i) {
    float v = acc[i] * BN_SCALE_C;
    orow[(size_t)i * Nout] = v > 0.0f ? v : 0.0f;
  }
}

// ---------------------------------------------------------------------------
// 3. Farthest-point sampling. One workgroup (256 threads) per batch; global
//    dist array; LDS tree argmax (tie -> lowest index, matching jnp.argmax).
// ---------------------------------------------------------------------------
__global__ void fps_kernel(const float* __restrict__ xyz, int N, int S,
                           int* __restrict__ idx_out, float* __restrict__ dist) {
  int b = blockIdx.x, t = threadIdx.x;
  const float* X = xyz + (size_t)b * N * 3;
  float* D = dist + (size_t)b * N;
  __shared__ float sv[256];
  __shared__ int   si[256];
  __shared__ float lp[3];
  for (int i = t; i < N; i += 256) D[i] = 1e10f;
  if (t == 0) idx_out[b * S] = 0;
  int last = 0;
  __syncthreads();
  for (int s = 1; s < S; ++s) {
    if (t == 0) { lp[0] = X[last*3]; lp[1] = X[last*3+1]; lp[2] = X[last*3+2]; }
    __syncthreads();
    float bv = -1.0f; int bi = 0;
    for (int i = t; i < N; i += 256) {
      float dx = X[i*3] - lp[0], dy = X[i*3+1] - lp[1], dz = X[i*3+2] - lp[2];
      float d = dx*dx + dy*dy + dz*dz;
      float dn = fminf(D[i], d);
      D[i] = dn;
      if (dn > bv) { bv = dn; bi = i; }
    }
    sv[t] = bv; si[t] = bi;
    __syncthreads();
    for (int o = 128; o > 0; o >>= 1) {
      if (t < o) {
        if (sv[t + o] > sv[t] || (sv[t + o] == sv[t] && si[t + o] < si[t])) {
          sv[t] = sv[t + o]; si[t] = si[t + o];
        }
      }
      __syncthreads();
    }
    last = si[0];
    if (t == 0) idx_out[b * S + s] = last;
    __syncthreads();
  }
}

// ---------------------------------------------------------------------------
// 4. Ball query (first 32 in-radius indices, index order == reference's
//    smallest-index top_k) + grouping into padded WMMA A rows:
//    row = [xyz - center (3) | feats (C) | zeros -> Kp]
// ---------------------------------------------------------------------------
__global__ void group_kernel(const float* __restrict__ xyz, const float* __restrict__ feats,
                             const int* __restrict__ sidx, float* __restrict__ new_xyz,
                             float* __restrict__ G, int N, int S, int C, int Kp, float r2) {
  int q = blockIdx.x * blockDim.x + threadIdx.x;           // B*S queries
  if (q >= 4 * S) return;
  int b = q / S;
  const float* X = xyz + (size_t)b * N * 3;
  int ci = sidx[q];
  float cx = X[ci*3], cy = X[ci*3+1], cz = X[ci*3+2];
  new_xyz[(size_t)q * 3 + 0] = cx;
  new_xyz[(size_t)q * 3 + 1] = cy;
  new_xyz[(size_t)q * 3 + 2] = cz;
  int nidx[32]; int cnt = 0;
  for (int i = 0; i < N && cnt < 32; ++i) {
    float dx = X[i*3] - cx, dy = X[i*3+1] - cy, dz = X[i*3+2] - cz;
    if (dx*dx + dy*dy + dz*dz < r2) nidx[cnt++] = i;
  }
  int fill = (cnt > 0) ? nidx[0] : 0;
  for (int j = cnt; j < 32; ++j) nidx[j] = fill;
  const float* Fe = feats + (size_t)b * N * C;
  float* rowBase = G + (size_t)q * 32 * Kp;
  for (int j = 0; j < 32; ++j) {
    int p = nidx[j];
    float* r = rowBase + (size_t)j * Kp;
    r[0] = X[p*3] - cx; r[1] = X[p*3+1] - cy; r[2] = X[p*3+2] - cz;
    const float* fp = Fe + (size_t)p * C;
    for (int c = 0; c < C; ++c) r[3 + c] = fp[c];
    for (int c = 3 + C; c < Kp; ++c) r[c] = 0.0f;
  }
}

// ---------------------------------------------------------------------------
// 5. Max-pool over the 32 grouped samples: (B*S*32, C) -> (B*S, C)
// ---------------------------------------------------------------------------
__global__ void maxpool_kernel(const float* __restrict__ G, float* __restrict__ out,
                               int S, int C) {
  int i = blockIdx.x * blockDim.x + threadIdx.x;           // B*S*C
  if (i >= 4 * S * C) return;
  int c = i % C, bs = i / C;
  const float* base = G + (size_t)bs * 32 * C + c;
  float m = base[0];
  for (int j = 1; j < 32; ++j) m = fmaxf(m, base[(size_t)j * C]);
  out[i] = m;
}

// ---------------------------------------------------------------------------
// 6. FP: 3-NN inverse-distance interpolation + concat -> padded WMMA A rows
// ---------------------------------------------------------------------------
__global__ void interp_kernel(const float* __restrict__ uxyz, const float* __restrict__ kxyz,
                              const float* __restrict__ kfeat, const float* __restrict__ ufeat,
                              float* __restrict__ G, int n, int m, int Ck, int Cu, int Kp) {
  int q = blockIdx.x * blockDim.x + threadIdx.x;           // B*n
  if (q >= 4 * n) return;
  int b = q / n, i = q % n;
  const float* U = uxyz + (size_t)b * n * 3;
  const float* K = kxyz + (size_t)b * m * 3;
  float px = U[i*3], py = U[i*3+1], pz = U[i*3+2];
  float d0 = 3.4e38f, d1 = 3.4e38f, d2 = 3.4e38f; int i0 = 0, i1 = 0, i2 = 0;
  for (int j = 0; j < m; ++j) {
    float dx = K[j*3] - px, dy = K[j*3+1] - py, dz = K[j*3+2] - pz;
    float d = dx*dx + dy*dy + dz*dz;
    if (d < d0)      { d2 = d1; i2 = i1; d1 = d0; i1 = i0; d0 = d; i0 = j; }
    else if (d < d1) { d2 = d1; i2 = i1; d1 = d;  i1 = j; }
    else if (d < d2) { d2 = d;  i2 = j; }
  }
  float w0 = 1.0f / (d0 + 1e-8f), w1 = 1.0f / (d1 + 1e-8f), w2 = 1.0f / (d2 + 1e-8f);
  float ws = w0 + w1 + w2; w0 /= ws; w1 /= ws; w2 /= ws;
  const float* F = kfeat + (size_t)b * m * Ck;
  const float* f0 = F + (size_t)i0 * Ck;
  const float* f1 = F + (size_t)i1 * Ck;
  const float* f2 = F + (size_t)i2 * Ck;
  const float* fu = ufeat + ((size_t)b * n + i) * Cu;
  float* r = G + (size_t)q * Kp;
  for (int c = 0; c < Ck; ++c) r[c] = w0 * f0[c] + w1 * f1[c] + w2 * f2[c];
  for (int c = 0; c < Cu; ++c) r[Ck + c] = fu[c];
  for (int c = Ck + Cu; c < Kp; ++c) r[c] = 0.0f;
}

// ---------------------------------------------------------------------------
// 7. Head: exact FP0 (point 0 only) + fc + control-point MLP + sigmoid.
//    One block of 160 threads, loops over the 4 batches. ~0.7 MFLOP total.
// ---------------------------------------------------------------------------
__global__ void head_kernel(const float* __restrict__ xyz0, const float* __restrict__ feats0,
                            const float* __restrict__ xyz1, const float* __restrict__ f1n,
                            const float* __restrict__ w0a, const float* __restrict__ w0b,
                            const float* __restrict__ w0c, const float* __restrict__ fcw,
                            const float* __restrict__ cp,  const float* __restrict__ cpw,
                            const float* __restrict__ cpb, const float* __restrict__ asw1,
                            const float* __restrict__ asb1,const float* __restrict__ asw2,
                            const float* __restrict__ asb2,float* __restrict__ out) {
  __shared__ float vin[160];
  __shared__ float h1[128];
  __shared__ float h2[128];
  __shared__ int   nn[3];
  __shared__ float nw[3];
  int t = threadIdx.x;
  for (int b = 0; b < 4; ++b) {
    if (t == 0) {                       // 3-NN of point 0 among 1024 level-1 pts
      const float* U = xyz0 + (size_t)b * 16384 * 3;
      const float* K = xyz1 + (size_t)b * 1024 * 3;
      float px = U[0], py = U[1], pz = U[2];
      float d0 = 3.4e38f, d1 = 3.4e38f, d2 = 3.4e38f; int i0 = 0, i1 = 0, i2 = 0;
      for (int j = 0; j < 1024; ++j) {
        float dx = K[j*3] - px, dy = K[j*3+1] - py, dz = K[j*3+2] - pz;
        float d = dx*dx + dy*dy + dz*dz;
        if (d < d0)      { d2 = d1; i2 = i1; d1 = d0; i1 = i0; d0 = d; i0 = j; }
        else if (d < d1) { d2 = d1; i2 = i1; d1 = d;  i1 = j; }
        else if (d < d2) { d2 = d;  i2 = j; }
      }
      float w0 = 1.0f / (d0 + 1e-8f), w1 = 1.0f / (d1 + 1e-8f), w2 = 1.0f / (d2 + 1e-8f);
      float ws = w0 + w1 + w2;
      nn[0] = i0; nn[1] = i1; nn[2] = i2;
      nw[0] = w0 / ws; nw[1] = w1 / ws; nw[2] = w2 / ws;
    }
    __syncthreads();
    if (t < 128) {
      const float* F = f1n + (size_t)b * 1024 * 128;
      vin[t] = nw[0] * F[(size_t)nn[0]*128 + t] + nw[1] * F[(size_t)nn[1]*128 + t]
             + nw[2] * F[(size_t)nn[2]*128 + t];
    }
    if (t >= 128 && t < 132) vin[t] = feats0[(size_t)b * 16384 * 4 + (t - 128)];
    __syncthreads();
    if (t < 128) { float a = 0; for (int c = 0; c < 132; ++c) a += vin[c] * w0a[c*128 + t];
                   h1[t] = fmaxf(a * BN_SCALE_C, 0.0f); }
    __syncthreads();
    if (t < 128) { float a = 0; for (int c = 0; c < 128; ++c) a += h1[c] * w0b[c*128 + t];
                   h2[t] = fmaxf(a * BN_SCALE_C, 0.0f); }
    __syncthreads();
    if (t < 128) { float a = 0; for (int c = 0; c < 128; ++c) a += h2[c] * w0c[c*128 + t];
                   h1[t] = fmaxf(a * BN_SCALE_C, 0.0f); }
    __syncthreads();
    if (t < 128) { float a = 0; for (int c = 0; c < 128; ++c) a += h1[c] * fcw[c*128 + t];
                   h2[t] = fmaxf(a * BN_SCALE_C, 0.0f); }   // net
    __syncthreads();
    if (t < 128) vin[t] = h2[t];
    if (t < 32)  { float a = cpb[t]; for (int k = 0; k < 3; ++k) a += cp[b*3 + k] * cpw[k*32 + t];
                   vin[128 + t] = a; }
    __syncthreads();
    if (t < 128) { float a = asb1[t]; for (int c = 0; c < 160; ++c) a += vin[c] * asw1[c*128 + t];
                   h1[t] = (a > 0.0f) ? a : 0.01f * a; }
    __syncthreads();
    if (t == 0)  { float a = asb2[0]; for (int c = 0; c < 128; ++c) a += h1[c] * asw2[c];
                   out[b] = 1.0f / (1.0f + expf(-a)); }
    __syncthreads();
  }
}

// ---------------------------------------------------------------------------
// Host-side orchestration
// ---------------------------------------------------------------------------
static inline void launch_gemm(const float* A, const _Float16* Wf, float* Out,
                               int M, int Kp, int Nout, hipStream_t s) {
  int tiles = (M >> 4) * (Nout >> 4);
  int blocks = (tiles + 7) >> 3;
  gemm_wmma_relu<<<blocks, 256, 0, s>>>(A, Wf, Out, M, Kp, Nout);
}

struct LayerW { int din; int Cin; int Cout; int Kp; };

extern "C" void kernel_launch(void* const* d_in, const int* in_sizes, int n_in,
                              void* d_out, int out_size, void* d_ws, size_t ws_size,
                              hipStream_t stream) {
  (void)in_sizes; (void)n_in; (void)out_size; (void)ws_size;
  float* ws = (float*)d_ws;
  size_t o = 0;
  auto alloc = [&](size_t n) { size_t r = o; o += n; return r; };

  // workspace layout (floats)
  size_t xyz0_o  = alloc(4 * 16384 * 3);
  size_t feats0_o= alloc(4 * 16384 * 4);
  size_t dist_o  = alloc(4 * 16384);
  size_t idx_o   = alloc(8192);                     // int region
  size_t xyz1_o  = alloc(4 * 1024 * 3);
  size_t f1_o    = alloc(4 * 1024 * 64);
  size_t xyz2_o  = alloc(4 * 256 * 3);
  size_t f2_o    = alloc(4 * 256 * 128);
  size_t xyz3_o  = alloc(4 * 64 * 3);
  size_t f3_o    = alloc(4 * 64 * 256);
  size_t xyz4_o  = alloc(4 * 16 * 3);
  size_t f4_o    = alloc(4 * 16 * 512);
  size_t f3n_o   = alloc(4 * 64 * 256);
  size_t f2n_o   = alloc(4 * 256 * 256);
  size_t f1n_o   = alloc(4 * 1024 * 128);
  size_t bufA_o  = alloc(8388608);                  // 131072 x 64 max
  size_t bufB_o  = alloc(8388608);
  size_t wfrag_o = alloc(460800);                   // 903168 halves + slack

  const float* pcs = (const float*)d_in[0];
  const float* cp  = (const float*)d_in[1];

  // 0. prep
  prep_kernel<<<(4 * 16384 + 255) / 256, 256, 0, stream>>>(pcs, cp, ws + xyz0_o, ws + feats0_o);

  // 1. weight fragment conversion (18 WMMA layers)
  static const LayerW LW[18] = {
    {2,   7,  32,  32}, {3,  32,  32,  32}, {4,  32,  64,  32},   // SA1
    {5,  67,  64,  96}, {6,  64,  64,  64}, {7,  64, 128,  64},   // SA2
    {8, 131, 128, 160}, {9, 128, 128, 128}, {10,128, 256, 128},   // SA3
    {11,259, 256, 288}, {12,256, 256, 256}, {13,256, 512, 256},   // SA4
    {21,768, 256, 768}, {22,256, 256, 256},                       // FP3
    {19,384, 256, 384}, {20,256, 256, 256},                       // FP2
    {17,320, 256, 320}, {18,256, 128, 256},                       // FP1
  };
  _Float16* wf = (_Float16*)(ws + wfrag_o);
  size_t wo[18]; size_t cur = 0;
  for (int i = 0; i < 18; ++i) { wo[i] = cur; cur += (size_t)LW[i].Kp * LW[i].Cout; }
  for (int i = 0; i < 18; ++i) {
    int total = LW[i].Kp * LW[i].Cout;
    wfrag_kernel<<<(total + 255) / 256, 256, 0, stream>>>(
        (const float*)d_in[LW[i].din], wf + wo[i], LW[i].Cin, LW[i].Cout, LW[i].Kp);
  }

  // 2. SA levels
  int   saN[4]  = {16384, 1024, 256, 64};
  int   saS[4]  = {1024, 256, 64, 16};
  int   saC[4]  = {4, 64, 128, 256};
  int   saKp[4] = {32, 96, 160, 288};
  float saR[4]  = {0.1f, 0.2f, 0.4f, 0.8f};
  int   saO1[4] = {32, 64, 128, 256};
  int   saO2[4] = {32, 64, 128, 256};
  int   saO3[4] = {64, 128, 256, 512};
  size_t xyzIn[4]  = {xyz0_o, xyz1_o, xyz2_o, xyz3_o};
  size_t featIn[4] = {feats0_o, f1_o, f2_o, f3_o};
  size_t xyzOut[4] = {xyz1_o, xyz2_o, xyz3_o, xyz4_o};
  size_t featOut[4]= {f1_o, f2_o, f3_o, f4_o};
  int* ibase = (int*)(ws + idx_o);
  int* idxPtr[4] = {ibase, ibase + 4096, ibase + 5120, ibase + 5376};

  for (int l = 0; l < 4; ++l) {
    fps_kernel<<<4, 256, 0, stream>>>(ws + xyzIn[l], saN[l], saS[l], idxPtr[l], ws + dist_o);
    int Q = 4 * saS[l];
    group_kernel<<<(Q + 63) / 64, 64, 0, stream>>>(
        ws + xyzIn[l], ws + featIn[l], idxPtr[l], ws + xyzOut[l], ws + bufA_o,
        saN[l], saS[l], saC[l], saKp[l], saR[l] * saR[l]);
    int M = Q * 32;
    launch_gemm(ws + bufA_o, wf + wo[3*l+0], ws + bufB_o, M, saKp[l], saO1[l], stream);
    launch_gemm(ws + bufB_o, wf + wo[3*l+1], ws + bufA_o, M, saO1[l], saO2[l], stream);
    launch_gemm(ws + bufA_o, wf + wo[3*l+2], ws + bufB_o, M, saO2[l], saO3[l], stream);
    int tot = Q * saO3[l];
    maxpool_kernel<<<(tot + 255) / 256, 256, 0, stream>>>(ws + bufB_o, ws + featOut[l],
                                                          saS[l], saO3[l]);
  }

  // 3. FP3: unk=level3(64), kn=level4(16): 768 -> 256 -> 256
  interp_kernel<<<(4 * 64 + 63) / 64, 64, 0, stream>>>(
      ws + xyz3_o, ws + xyz4_o, ws + f4_o, ws + f3_o, ws + bufA_o, 64, 16, 512, 256, 768);
  launch_gemm(ws + bufA_o, wf + wo[12], ws + bufB_o, 256, 768, 256, stream);
  launch_gemm(ws + bufB_o, wf + wo[13], ws + f3n_o, 256, 256, 256, stream);

  // FP2: unk=level2(256), kn=level3(64): 384 -> 256 -> 256
  interp_kernel<<<(4 * 256 + 63) / 64, 64, 0, stream>>>(
      ws + xyz2_o, ws + xyz3_o, ws + f3n_o, ws + f2_o, ws + bufA_o, 256, 64, 256, 128, 384);
  launch_gemm(ws + bufA_o, wf + wo[14], ws + bufB_o, 1024, 384, 256, stream);
  launch_gemm(ws + bufB_o, wf + wo[15], ws + f2n_o, 1024, 256, 256, stream);

  // FP1: unk=level1(1024), kn=level2(256): 320 -> 256 -> 128
  interp_kernel<<<(4 * 1024 + 63) / 64, 64, 0, stream>>>(
      ws + xyz1_o, ws + xyz2_o, ws + f2n_o, ws + f1_o, ws + bufA_o, 1024, 256, 256, 64, 320);
  launch_gemm(ws + bufA_o, wf + wo[16], ws + bufB_o, 4096, 320, 256, stream);
  launch_gemm(ws + bufB_o, wf + wo[17], ws + f1n_o, 4096, 256, 128, stream);

  // 4. head (exact FP0 at point 0 + fc + attention head + sigmoid)
  head_kernel<<<1, 160, 0, stream>>>(
      ws + xyz0_o, ws + feats0_o, ws + xyz1_o, ws + f1n_o,
      (const float*)d_in[14], (const float*)d_in[15], (const float*)d_in[16],
      (const float*)d_in[23], cp, (const float*)d_in[24], (const float*)d_in[25],
      (const float*)d_in[26], (const float*)d_in[27], (const float*)d_in[28],
      (const float*)d_in[29], (float*)d_out);
}
